// GraphSAGE_42296837931009
// MI455X (gfx1250) — matrix operand order repaired and checked by reading, not compiled, not run on hardware
//
#include <hip/hip_runtime.h>

typedef __attribute__((ext_vector_type(16))) _Float16 v16h;
typedef __attribute__((ext_vector_type(8)))  float    v8f;

#define NNODES 50000
#define NEDGES 800000
#define FIN    1433

// ---------------------------------------------------------------------------
// zero fill
// ---------------------------------------------------------------------------
__global__ void zero_f32(float* __restrict__ p, long n) {
  long i = (long)blockIdx.x * blockDim.x + threadIdx.x;
  if (i < n) p[i] = 0.0f;
}

// pack two f32 into one dword of two f16 with a single v_cvt_pk_rtz_f16_f32
__device__ __forceinline__ unsigned pk2h(float a, float b) {
  return __builtin_bit_cast(unsigned, __builtin_amdgcn_cvt_pkrtz(a, b));
}

// ---------------------------------------------------------------------------
// GEMM1: XW[50000x128] = X[50000x1433] @ [W1_l | W1_r][1433x128]
// f16 WMMA, f32 accumulate. X and W tiles staged through double-buffered LDS
// (pair-packed ds_store_b32, conflict-free 80B row stride). One barrier per
// K-step; next tile's global loads + LDS commits overlap current WMMAs.
// Block = 256 threads = 8 waves; wave w owns rows m0+16w..+15, all 128 cols.
// ---------------------------------------------------------------------------
__global__ __launch_bounds__(256) void sage_gemm1(
    const float* __restrict__ x, const float* __restrict__ Wl,
    const float* __restrict__ Wr, float* __restrict__ XW) {
  __shared__ _Float16 Xt[2][128][40];        // [buf][row][k], 80B stride
  __shared__ _Float16 Wt[2][128][40];        // [buf][n][k]

  const int tid  = threadIdx.x;
  const int lane = tid & 31;
  const int wave = tid >> 5;
  const int hi   = (lane & 16) ? 1 : 0;      // lane half selects K sub-block
  const int nl   = lane & 15;
  const int m0   = blockIdx.x * 128 + wave * 16;
  const int gr0  = blockIdx.x * 128;

  // X tile pair map: idx = tid + i*256 -> row = idx>>4, kpair = idx&15
  const int xr_r  = tid >> 4;                // row for i=0; +16 per i
  const int xr_kp = tid & 15;                // k pair index
  // W tile pair map: idx = tid + i*256 -> n = idx&127, kpair = idx>>7
  const int wn    = tid & 127;
  const int wkp0  = tid >> 7;                // kpair for i=0; +2 per i

  v8f acc[8];
#pragma unroll
  for (int j = 0; j < 8; ++j)
#pragma unroll
    for (int i = 0; i < 8; ++i) acc[j][i] = 0.0f;

  float xa[8], xb[8], wa[8], wb[8];

  auto load_x = [&](int k0) {
#pragma unroll
    for (int i = 0; i < 8; ++i) {
      int r  = xr_r + 16 * i;
      int gr = gr0 + r; if (gr >= NNODES) gr = NNODES - 1;
      int k  = k0 + 2 * xr_kp;
      long base = (long)gr * FIN + k;
      xa[i] = (k     < FIN) ? x[base]     : 0.0f;
      xb[i] = (k + 1 < FIN) ? x[base + 1] : 0.0f;
    }
  };
  auto load_w = [&](int k0) {
#pragma unroll
    for (int i = 0; i < 8; ++i) {
      int k = k0 + 2 * (wkp0 + 2 * i);
      float a = 0.0f, b = 0.0f;
      if (k < FIN)
        a = (wn < 64) ? Wl[(long)k * 64 + wn] : Wr[(long)k * 64 + (wn - 64)];
      if (k + 1 < FIN)
        b = (wn < 64) ? Wl[(long)(k + 1) * 64 + wn]
                      : Wr[(long)(k + 1) * 64 + (wn - 64)];
      wa[i] = a; wb[i] = b;
    }
  };
  auto commit = [&](int buf) {
#pragma unroll
    for (int i = 0; i < 8; ++i)
      *(unsigned*)&Xt[buf][xr_r + 16 * i][2 * xr_kp] = pk2h(xa[i], xb[i]);
#pragma unroll
    for (int i = 0; i < 8; ++i)
      *(unsigned*)&Wt[buf][wn][2 * (wkp0 + 2 * i)] = pk2h(wa[i], wb[i]);
  };

  // prologue: tile 0 -> buffer 0
  load_x(0); load_w(0);
  commit(0);
  __syncthreads();

  int cur = 0;
  for (int k0 = 0; k0 < FIN; k0 += 32) {
    const int kn = k0 + 32;
    const bool more = kn < FIN;
    if (more) { load_x(kn); load_w(kn); }    // global loads in flight

    // A fragment: 2 x ds_load_b128 (ISA 16-bit A 16x32 layout)
    union { uint4 u[2]; v16h h; } aa;
    const _Float16* pa = &Xt[cur][wave * 16 + nl][hi * 8];
    aa.u[0] = *(const uint4*)(pa);           // K = hi*8 .. hi*8+7
    aa.u[1] = *(const uint4*)(pa + 16);      // K = 16+hi*8 ..

    const int koff = hi * 8;
#pragma unroll
    for (int j = 0; j < 8; ++j) {
      union { uint4 u[2]; v16h h; } bb;
      const _Float16* p = &Wt[cur][j * 16 + nl][koff];
      bb.u[0] = *(const uint4*)(p);
      bb.u[1] = *(const uint4*)(p + 16);
      acc[j] = __builtin_amdgcn_wmma_f32_16x16x32_f16(
          false, aa.h, false, bb.h, (short)0, acc[j], false, false);
    }

    if (more) commit(cur ^ 1);               // waits land after WMMA issue
    __syncthreads();
    cur ^= 1;
  }

#pragma unroll
  for (int j = 0; j < 8; ++j)
#pragma unroll
    for (int v = 0; v < 8; ++v) {
      int rr = m0 + v + hi * 8;              // C/D layout: VGPR v -> M=v(+8)
      if (rr < NNODES) XW[(long)rr * 128 + j * 16 + nl] = acc[j][v];
    }
}

// ---------------------------------------------------------------------------
// edge scatter 1: agg1[dst] += XWl[src] (cols 0..63), deg[dst] += 1
// ---------------------------------------------------------------------------
__global__ __launch_bounds__(256) void sage_scatter1(
    const int* __restrict__ ei, const float* __restrict__ XW,
    float* __restrict__ agg1, float* __restrict__ deg) {
  const int lane = threadIdx.x & 31;
  const int e = blockIdx.x * 8 + (threadIdx.x >> 5);
  if (e >= NEDGES) return;
  const int src = ei[e];
  const int dst = ei[NEDGES + e];
  const float* p = XW + (long)src * 128;     // left half = X @ W1_l
  atomicAdd(&agg1[(long)dst * 64 + lane],      p[lane]);
  atomicAdd(&agg1[(long)dst * 64 + lane + 32], p[lane + 32]);
  if (lane == 0) atomicAdd(&deg[dst], 1.0f);
}

// ---------------------------------------------------------------------------
// combine1: h = agg1/deg + b1 + XWr, stored as f16 for layer-2 WMMA
// ---------------------------------------------------------------------------
__global__ __launch_bounds__(256) void sage_combine1(
    const float* __restrict__ XW, const float* __restrict__ agg1,
    const float* __restrict__ deg, const float* __restrict__ b1,
    _Float16* __restrict__ H) {
  long i = (long)blockIdx.x * 256 + threadIdx.x;
  if (i >= (long)NNODES * 64) return;
  int r = (int)(i >> 6), f = (int)(i & 63);
  float d = deg[r];
  float a = d > 0.0f ? agg1[i] / d : 0.0f;
  H[i] = (_Float16)(a + b1[f] + XW[(long)r * 128 + 64 + f]);
}

// ---------------------------------------------------------------------------
// GEMM2: HW[50000x64] = H[50000x64](f16) @ [W2_l | W2_r][64x64]
// whole W2 lives in LDS; 2 K-steps of 32; 4 N-tiles per wave
// ---------------------------------------------------------------------------
__global__ __launch_bounds__(256) void sage_gemm2(
    const _Float16* __restrict__ H, const float* __restrict__ Wl,
    const float* __restrict__ Wr, float* __restrict__ HW) {
  __shared__ _Float16 Wt[64][72];            // [n][k], stride 144B
  const int lane = threadIdx.x & 31;
  const int wave = threadIdx.x >> 5;
  const int hi   = (lane & 16) ? 1 : 0;
  const int nl   = lane & 15;
  const int m0   = blockIdx.x * 128 + wave * 16;
  const int row  = m0 + nl;
  const long rbase = (long)(row < NNODES ? row : NNODES - 1) * 64;

  // pair-packed staging: idx = tid + i*256 -> n = idx&63, kpair = idx>>6
  {
    const int n   = threadIdx.x & 63;
    const int kp0 = threadIdx.x >> 6;        // +4 per i
#pragma unroll
    for (int i = 0; i < 8; ++i) {
      int k = 2 * (kp0 + 4 * i);
      float a = (n < 32) ? Wl[k * 32 + n]       : Wr[k * 32 + (n - 32)];
      float b = (n < 32) ? Wl[(k + 1) * 32 + n] : Wr[(k + 1) * 32 + (n - 32)];
      *(unsigned*)&Wt[n][k] = pk2h(a, b);
    }
  }
  __syncthreads();

  v8f acc[4];
#pragma unroll
  for (int j = 0; j < 4; ++j)
#pragma unroll
    for (int i = 0; i < 8; ++i) acc[j][i] = 0.0f;

#pragma unroll
  for (int k0 = 0; k0 < 64; k0 += 32) {
    union { uint4 u[2]; v16h h; } aa;        // H already f16: b128 loads
    const _Float16* pa = H + rbase + k0 + hi * 8;
    aa.u[0] = *(const uint4*)(pa);
    aa.u[1] = *(const uint4*)(pa + 16);
    const int koff = k0 + hi * 8;
#pragma unroll
    for (int j = 0; j < 4; ++j) {
      union { uint4 u[2]; v16h h; } bb;
      const _Float16* p = &Wt[j * 16 + nl][koff];
      bb.u[0] = *(const uint4*)(p);
      bb.u[1] = *(const uint4*)(p + 16);
      acc[j] = __builtin_amdgcn_wmma_f32_16x16x32_f16(
          false, aa.h, false, bb.h, (short)0, acc[j], false, false);
    }
  }

#pragma unroll
  for (int j = 0; j < 4; ++j)
#pragma unroll
    for (int v = 0; v < 8; ++v) {
      int rr = m0 + v + hi * 8;
      if (rr < NNODES) HW[(long)rr * 64 + j * 16 + nl] = acc[j][v];
    }
}

// ---------------------------------------------------------------------------
// edge scatter 2: agg2[dst] += HWl[src] (cols 0..31)
// ---------------------------------------------------------------------------
__global__ __launch_bounds__(256) void sage_scatter2(
    const int* __restrict__ ei, const float* __restrict__ HW,
    float* __restrict__ agg2) {
  const int lane = threadIdx.x & 31;
  const int e = blockIdx.x * 8 + (threadIdx.x >> 5);
  if (e >= NEDGES) return;
  const int src = ei[e];
  const int dst = ei[NEDGES + e];
  atomicAdd(&agg2[(long)dst * 32 + lane], HW[(long)src * 64 + lane]);
}

// ---------------------------------------------------------------------------
// final: relu(agg2/deg + b2 + HWr) then log_softmax over 32 cols (one wave/row)
// ---------------------------------------------------------------------------
__global__ __launch_bounds__(256) void sage_final(
    const float* __restrict__ HW, const float* __restrict__ agg2,
    const float* __restrict__ deg, const float* __restrict__ b2,
    float* __restrict__ out) {
  const int lane = threadIdx.x & 31;
  const int r = blockIdx.x * 8 + (threadIdx.x >> 5);
  if (r >= NNODES) return;
  float d = deg[r];
  float a = d > 0.0f ? agg2[(long)r * 32 + lane] / d : 0.0f;
  float v = a + b2[lane] + HW[(long)r * 64 + 32 + lane];
  v = fmaxf(v, 0.0f);
  float m = v;
#pragma unroll
  for (int off = 16; off; off >>= 1) m = fmaxf(m, __shfl_xor(m, off, 32));
  float s = __expf(v - m);
#pragma unroll
  for (int off = 16; off; off >>= 1) s += __shfl_xor(s, off, 32);
  out[(long)r * 32 + lane] = v - m - __logf(s);
}

// ---------------------------------------------------------------------------
// launch
// ---------------------------------------------------------------------------
extern "C" void kernel_launch(void* const* d_in, const int* in_sizes, int n_in,
                              void* d_out, int out_size, void* d_ws, size_t ws_size,
                              hipStream_t stream) {
  (void)in_sizes; (void)n_in; (void)out_size; (void)ws_size;
  const float* x   = (const float*)d_in[0];
  const int*   ei  = (const int*)d_in[1];
  const float* W1l = (const float*)d_in[2];
  const float* b1  = (const float*)d_in[3];
  const float* W1r = (const float*)d_in[4];
  const float* W2l = (const float*)d_in[5];
  const float* b2  = (const float*)d_in[6];
  const float* W2r = (const float*)d_in[7];
  float* out = (float*)d_out;

  // workspace layout (float offsets)
  float* ws   = (float*)d_ws;
  float* XW   = ws;                    // 50000*128 = 6,400,000
  float* agg1 = ws + 6400000;          // 3,200,000
  float* agg2 = ws + 9600000;          // 1,600,000
  float* deg  = ws + 11200000;         //    50,000
  float* HW   = ws + 11250000;         // 3,200,000
  _Float16* H = (_Float16*)(ws + 14450000);  // 3,200,000 halves (16B aligned)

  // 1) zero agg1 | agg2 | deg (contiguous: 4,850,000 floats)
  {
    long n = 4850000;
    zero_f32<<<(unsigned)((n + 255) / 256), 256, 0, stream>>>(agg1, n);
  }
  // 2) layer-1 fused GEMM (WMMA)
  sage_gemm1<<<(NNODES + 127) / 128, 256, 0, stream>>>(x, W1l, W1r, XW);
  // 3) scatter + degree
  sage_scatter1<<<NEDGES / 8, 256, 0, stream>>>(ei, XW, agg1, deg);
  // 4) combine -> h (f16)
  sage_combine1<<<(NNODES * 64 + 255) / 256, 256, 0, stream>>>(XW, agg1, deg, b1, H);
  // 5) layer-2 fused GEMM (WMMA)
  sage_gemm2<<<(NNODES + 127) / 128, 256, 0, stream>>>(H, W2l, W2r, HW);
  // 6) scatter
  sage_scatter2<<<NEDGES / 8, 256, 0, stream>>>(ei, HW, agg2);
  // 7) relu + log_softmax
  sage_final<<<(NNODES + 7) / 8, 256, 0, stream>>>(HW, agg2, deg, b2, out);
}
